// Transformer_1477468750505
// MI455X (gfx1250) — matrix operand order repaired and checked by reading, not compiled, not run on hardware
//
#include <hip/hip_runtime.h>
#include <cstdint>
#include <cstddef>

// ---------------------------------------------------------------------------
// CDNA5 (gfx1250, wave32) types
// ---------------------------------------------------------------------------
typedef __attribute__((ext_vector_type(16))) __bf16 v16bf;
typedef __attribute__((ext_vector_type(8)))  float  v8f;
typedef __attribute__((ext_vector_type(8)))  short  short8;
typedef __attribute__((ext_vector_type(4)))  unsigned int u32x4;
typedef __attribute__((ext_vector_type(8)))  int    i32x8;
typedef __attribute__((ext_vector_type(4)))  int    i32x4;

union FragBF {
  v16bf  v;
  short8 s[2];
};

__device__ __forceinline__ short f32_to_bf16(float f) {
  union { float f; unsigned u; } cv;
  cv.f = f;
  unsigned u = cv.u;
  unsigned r = u + 0x7fffu + ((u >> 16) & 1u);   // round-to-nearest-even
  return (short)(r >> 16);
}

// ---------------------------------------------------------------------------
// Tensor Data Mover staging (CDNA5): 2D bf16 tile (32-elem rows) -> LDS.
// D# group1 word0 = 0x06D10000:
//   data_size=1 (2B), pad_enable=1, pad_interval=3 (16 DWORDs = one 64B row),
//   pad_amount=3 (4 DWORDs = 16B)  => LDS row pitch 80B (bank-conflict-free).
// tensor_dim1 = #valid rows => TDM zero-fills OOB tail rows for us.
// ---------------------------------------------------------------------------
#if defined(__gfx1250__) && __has_builtin(__builtin_amdgcn_tensor_load_to_lds)
#define USE_TDM 1
#else
#define USE_TDM 0
#endif

#if USE_TDM
__device__ __forceinline__ void tdm_load_2d_bf16(
    const short* gsrc, unsigned lds_byte_addr,
    int rows_valid, int tile_rows, int stride_elems)
{
  unsigned long long ga = (unsigned long long)(size_t)gsrc;
  int td1 = rows_valid < tile_rows ? rows_valid : tile_rows;

  u32x4 g0;
  g0[0] = 1u;                                        // count=1, normal load D#
  g0[1] = lds_byte_addr;                             // LDS dest (byte addr)
  g0[2] = (unsigned)(ga & 0xffffffffu);              // global_addr[31:0]
  g0[3] = (unsigned)((ga >> 32) & 0x01ffffffu)       // global_addr[56:32]
        | 0x80000000u;                               // type=2 ("image")

  unsigned td0 = 32u;                                // tensor dim0 == tile dim0
  unsigned s0  = (unsigned)stride_elems;             // dim0 stride (elements)
  i32x8 g1;
  g1[0] = (int)0x06D10000u;                          // see header comment
  g1[1] = (int)((td0 & 0xffffu) << 16);              // tensor_dim0[15:0]
  g1[2] = (int)(((td0 >> 16) & 0xffffu) | (((unsigned)td1 & 0xffffu) << 16));
  g1[3] = (int)((((unsigned)td1 >> 16) & 0xffffu) | (32u << 16));  // tile_dim0=32
  g1[4] = (int)((unsigned)tile_rows & 0xffffu);      // tile_dim1 (tile_dim2=0)
  g1[5] = (int)s0;                                   // tensor_dim0_stride[31:0]
  g1[6] = 0;                                         // stride hi, dim1_stride lo
  g1[7] = 0;
  i32x4 z4 = {0, 0, 0, 0};
#if __clang_major__ >= 23
  i32x8 z8 = {0, 0, 0, 0, 0, 0, 0, 0};
  __builtin_amdgcn_tensor_load_to_lds(g0, g1, z4, z4, z8, 0);
#else
  __builtin_amdgcn_tensor_load_to_lds(g0, g1, z4, z4, 0);
#endif
}
#endif

// ---------------------------------------------------------------------------
// bf16 NT GEMM: C = alpha * (A @ B^T) + bias [+relu]
//   A: M x K bf16 (lda), B: N x K bf16 (ldb)  -- both row-major
//   Cf: optional f32 out (ldc); Cb: optional bf16 out (ldcb, cbT -> transposed)
// Batched: base += (z/inner)*Out + (z%inner)*In   (element strides)
// Tile 256x128x32, 8 waves (4Mx2N), wave tile 64x64 => 16 WMMA / wave / k-step.
// LDS double-buffered; staged by TDM (async, 1 issuing wave) or sync fallback.
// ---------------------------------------------------------------------------
#define BM 256
#define BN 128
#define LDL 40        // LDS row pitch in shorts (80B)

__global__ __launch_bounds__(256) void gemm_bf16nt_wmma_kernel(
    const short* __restrict__ A, long long aOut, long long aIn, int lda,
    const short* __restrict__ B, long long bOut, long long bIn, int ldb,
    float* __restrict__ Cf, short* __restrict__ Cb, int cbT, int ldcb,
    long long cOut, long long cIn, int ldc,
    const float* __restrict__ bias,
    int M, int N, int K, int innerCnt, int doRelu, float alpha)
{
  int z  = blockIdx.z;
  int zo = z / innerCnt, zi = z % innerCnt;
  A += (long long)zo * aOut + (long long)zi * aIn;
  B += (long long)zo * bOut + (long long)zi * bIn;
  long long cBase = (long long)zo * cOut + (long long)zi * cIn;

  __shared__ short lA[2][BM * LDL];
  __shared__ short lB[2][BN * LDL];

  int n0   = blockIdx.x * BN;
  int m0   = blockIdx.y * BM;
  int tid  = threadIdx.x;
  int lane = tid & 31;
  int wave = tid >> 5;
  int wm   = wave & 3;                      // M offset wm*64
  int wn   = wave >> 2;                     // N offset wn*64
  int hh   = lane >> 4;
  int l16  = lane & 15;

  v8f acc[4][4];
  v8f vz = {0.f, 0.f, 0.f, 0.f, 0.f, 0.f, 0.f, 0.f};
#pragma unroll
  for (int i = 0; i < 4; ++i)
#pragma unroll
    for (int j = 0; j < 4; ++j) acc[i][j] = vz;

#if USE_TDM
  if (wave == 0) {
    tdm_load_2d_bf16(A + (long long)m0 * lda, (unsigned)(size_t)&lA[0][0],
                     M - m0, BM, lda);
    tdm_load_2d_bf16(B + (long long)n0 * ldb, (unsigned)(size_t)&lB[0][0],
                     N - n0, BN, ldb);
  }
#endif

  for (int k0 = 0; k0 < K; k0 += 32) {
    int cur = (k0 >> 5) & 1;

#if USE_TDM
    if (wave == 0) {
      int k1 = k0 + 32;
      if (k1 < K) {
        int nxt = cur ^ 1;
        tdm_load_2d_bf16(A + (long long)m0 * lda + k1,
                         (unsigned)(size_t)&lA[nxt][0], M - m0, BM, lda);
        tdm_load_2d_bf16(B + (long long)n0 * ldb + k1,
                         (unsigned)(size_t)&lB[nxt][0], N - n0, BN, ldb);
        __builtin_amdgcn_s_wait_tensorcnt((short)2);   // current slice resident
      } else {
        __builtin_amdgcn_s_wait_tensorcnt((short)0);
      }
    }
    __syncthreads();
#else
    __syncthreads();
    // ---- synchronous staging fallback: straight bf16 row copies ----
#pragma unroll
    for (int c2 = 0; c2 < 4; ++c2) {                  // A: 1024 16B chunks
      int c   = tid + c2 * 256;
      int row = c >> 2;
      int off = (c & 3) * 8;
      int gm  = m0 + row;
      short8 val = {0, 0, 0, 0, 0, 0, 0, 0};
      if (gm < M)
        val = *reinterpret_cast<const short8*>(A + (long long)gm * lda + k0 + off);
      *reinterpret_cast<short8*>(&lA[cur][row * LDL + off]) = val;
    }
#pragma unroll
    for (int c2 = 0; c2 < 2; ++c2) {                  // B: 512 16B chunks
      int c   = tid + c2 * 256;
      int row = c >> 2;
      int off = (c & 3) * 8;
      int gn  = n0 + row;
      short8 val = {0, 0, 0, 0, 0, 0, 0, 0};
      if (gn < N)
        val = *reinterpret_cast<const short8*>(B + (long long)gn * ldb + k0 + off);
      *reinterpret_cast<short8*>(&lB[cur][row * LDL + off]) = val;
    }
    __syncthreads();
#endif

    // ---- fragments (contiguous 16B ds loads per ISA VGPR layouts) ----
    FragBF fa[4], fb[4];
#pragma unroll
    for (int mi = 0; mi < 4; ++mi) {
      const short* rp = &lA[cur][(wm * 64 + mi * 16 + l16) * LDL];
      fa[mi].s[0] = *reinterpret_cast<const short8*>(rp + 8 * hh);
      fa[mi].s[1] = *reinterpret_cast<const short8*>(rp + 16 + 8 * hh);
    }
#pragma unroll
    for (int nj = 0; nj < 4; ++nj) {
      const short* rp = &lB[cur][(wn * 64 + nj * 16 + l16) * LDL];
      fb[nj].s[0] = *reinterpret_cast<const short8*>(rp + 8 * hh);
      fb[nj].s[1] = *reinterpret_cast<const short8*>(rp + 16 + 8 * hh);
    }

#if defined(__gfx1250__)
#pragma unroll
    for (int mi = 0; mi < 4; ++mi)
#pragma unroll
      for (int nj = 0; nj < 4; ++nj)
        acc[mi][nj] = __builtin_amdgcn_wmma_f32_16x16x32_bf16(
            false, fa[mi].v, false, fb[nj].v, (short)0, acc[mi][nj],
            false, false);
#endif

#if USE_TDM
    __syncthreads();   // all reads of buf[cur] done before TDM re-fills it
#endif
  }

  // ---- epilogue: C/D layout VGPR r -> M = r + 8*(lane/16), N = lane%16 ----
#pragma unroll
  for (int mi = 0; mi < 4; ++mi) {
#pragma unroll
    for (int nj = 0; nj < 4; ++nj) {
      int gn = n0 + wn * 64 + nj * 16 + l16;
      if (gn < N) {
        float bv = bias ? bias[gn] : 0.f;
#pragma unroll
        for (int rr = 0; rr < 8; ++rr) {
          int gm = m0 + wm * 64 + mi * 16 + hh * 8 + rr;
          if (gm < M) {
            float val = alpha * acc[mi][nj][rr] + bv;
            if (doRelu) val = fmaxf(val, 0.f);
            if (Cf) Cf[cBase + (long long)gm * ldc + gn] = val;
            if (Cb) {
              if (cbT) Cb[cBase + (long long)gn * ldcb + gm] = f32_to_bf16(val);
              else     Cb[cBase + (long long)gm * ldcb + gn] = f32_to_bf16(val);
            }
          }
        }
      }
    }
  }
}

// ---------------------------------------------------------------------------
// Weight convert + transpose:  Wt[n][k] = bf16(W[k][n])   (32x32 LDS tiles)
// ---------------------------------------------------------------------------
__global__ __launch_bounds__(256) void wt_transpose_bf16_kernel(
    const float* __restrict__ W, short* __restrict__ Wt, int K, int N)
{
  __shared__ float tle[32][33];
  int n0 = blockIdx.x * 32;
  int k0 = blockIdx.y * 32;
  int tx = threadIdx.x & 31;
  int ty = threadIdx.x >> 5;                // 0..7
#pragma unroll
  for (int j = 0; j < 4; ++j) {
    int k = ty + j * 8;
    tle[k][tx] = W[(long long)(k0 + k) * N + (n0 + tx)];
  }
  __syncthreads();
#pragma unroll
  for (int j = 0; j < 4; ++j) {
    int n = ty + j * 8;
    Wt[(long long)(n0 + n) * K + (k0 + tx)] = f32_to_bf16(tle[tx][n]);
  }
}

// ---------------------------------------------------------------------------
// Embedding gather + PE; writes optional f32 copy + bf16 copy.
// ---------------------------------------------------------------------------
__global__ __launch_bounds__(256) void embed_pe_kernel(
    const int* __restrict__ tok, const float* __restrict__ emb,
    const float* __restrict__ pe, float* __restrict__ outF,
    short* __restrict__ outB, int S, int E)
{
  long long idx = (long long)blockIdx.x * blockDim.x + threadIdx.x;
  int e        = (int)(idx % E);
  long long bs = idx / E;
  int s        = (int)(bs % S);
  int t        = tok[bs];
  float v = emb[(long long)t * E + e] + pe[(long long)s * E + e];
  if (outF) outF[idx] = v;
  outB[idx] = f32_to_bf16(v);
}

// ---------------------------------------------------------------------------
// Masked softmax over f32 rows of Sk==1024; writes bf16 probs IN PLACE over
// the first half of each row's bytes (all reads precede reduction barriers).
// ---------------------------------------------------------------------------
__global__ __launch_bounds__(256) void softmax_mask_kernel(
    float* __restrict__ sc, const int* __restrict__ tok,
    int Sq, int Sk, int H, int causal)
{
  long long row = blockIdx.x;
  int q = (int)(row % Sq);
  int b = (int)(row / ((long long)Sq * H));
  float* p = sc + row * (long long)Sk;
  int t = threadIdx.x;
  __shared__ float red[256];

  float vals[4];
  float mx = -3.0e38f;
#pragma unroll
  for (int i = 0; i < 4; ++i) {
    int kc  = t + i * 256;
    float v = p[kc];
    bool ok = causal ? (kc <= q) : (tok[(long long)b * Sk + kc] != 0);
    v       = ok ? v : -1.0e9f;
    vals[i] = v;
    mx      = fmaxf(mx, v);
  }
  red[t] = mx; __syncthreads();
  for (int s = 128; s > 0; s >>= 1) {
    if (t < s) red[t] = fmaxf(red[t], red[t + s]);
    __syncthreads();
  }
  mx = red[0]; __syncthreads();

  float sum = 0.f;
#pragma unroll
  for (int i = 0; i < 4; ++i) { vals[i] = __expf(vals[i] - mx); sum += vals[i]; }
  red[t] = sum; __syncthreads();
  for (int s = 128; s > 0; s >>= 1) {
    if (t < s) red[t] += red[t + s];
    __syncthreads();
  }
  float inv = 1.0f / red[0];

  short* pb = (short*)p;                  // bf16 probs, row pitch 2*Sk shorts
#pragma unroll
  for (int i = 0; i < 4; ++i) pb[t + i * 256] = f32_to_bf16(vals[i] * inv);
}

// ---------------------------------------------------------------------------
// y = LayerNorm(x + r) * gs + gb  (rows of E=1024); f32 out optional, bf16 out
// ---------------------------------------------------------------------------
__global__ __launch_bounds__(256) void ln_residual_kernel(
    const float* __restrict__ x, const float* __restrict__ r,
    const float* __restrict__ gs, const float* __restrict__ gb,
    float* __restrict__ y, short* __restrict__ yb)
{
  const int E = 1024;
  long long row = blockIdx.x;
  const float* px = x + row * E;
  const float* pr = r + row * E;
  int t = threadIdx.x;
  __shared__ float red1[256];
  __shared__ float red2[256];

  float v[4]; float s1 = 0.f, s2 = 0.f;
#pragma unroll
  for (int i = 0; i < 4; ++i) {
    int c   = t + i * 256;
    float a = px[c] + pr[c];
    v[i] = a; s1 += a; s2 += a * a;
  }
  red1[t] = s1; red2[t] = s2; __syncthreads();
  for (int s = 128; s > 0; s >>= 1) {
    if (t < s) { red1[t] += red1[t + s]; red2[t] += red2[t + s]; }
    __syncthreads();
  }
  float mu   = red1[0] * (1.0f / E);
  float var  = red2[0] * (1.0f / E) - mu * mu;
  float rstd = rsqrtf(var + 1e-5f);
#pragma unroll
  for (int i = 0; i < 4; ++i) {
    int c = t + i * 256;
    float o = (v[i] - mu) * rstd * gs[c] + gb[c];
    if (y) y[row * E + c] = o;
    yb[row * E + c] = f32_to_bf16(o);
  }
}

// ---------------------------------------------------------------------------
// Host orchestration.  Input map (flattened setup_inputs order):
//   0 src, 1 trg, 2 enc_emb, 3 dec_emb, 4 pe,
//   5..36 encoder (dead), 37..62 dec layer 0 (dead), 63..88 dec layer 1 (LIVE),
//   89 fc_w, 90 fc_b.
// Live layer base DB=63: +0..7 sa{wq,wk,wv,wo}{w,b}, +8..15 ca{...},
//   +16..19 f1/f2 {w,b}, +20/21 ln1, +22/23 ln2, +24/25 ln3.
// ---------------------------------------------------------------------------
extern "C" void kernel_launch(void* const* d_in, const int* in_sizes, int n_in,
                              void* d_out, int out_size, void* d_ws, size_t ws_size,
                              hipStream_t stream)
{
  (void)in_sizes; (void)n_in; (void)out_size; (void)ws_size;
  const int S = 1024, E = 1024, Hh = 16, DK = 64, DF = 4096, V = 16000;
  const int DB = 63;
  const int TOKS = 4 * S;                               // B*S = 4096 rows
  const long long SL = (long long)TOKS * E;             // 4M elements
  const long long SE = (long long)S * E;
  const long long SS = (long long)S * S;
  const long long EEc = (long long)E * E;
  const size_t MB = 1u << 20;

  const int* src       = (const int*)d_in[0];
  const int* trg       = (const int*)d_in[1];
  const float* enc_emb = (const float*)d_in[2];
  const float* dec_emb = (const float*)d_in[3];
  const float* pe      = (const float*)d_in[4];
  auto P = [&](int i) { return (const float*)d_in[i]; };

  // ---- workspace layout: [scores f32 256MB][f32 acts 64MB][bf16 region] ----
  char* base = (char*)d_ws;
  float* scores = (float*)base;                         // 16*SL f32
  float* f32r   = (float*)(base + 256 * MB);
  float* trgx = f32r + 0 * SL;
  float* x1   = f32r + 1 * SL;
  float* x2   = f32r + 2 * SL;
  float* tmp  = f32r + 3 * SL;
  short* bfr  = (short*)(base + 320 * MB);
  short* srcxb = bfr + 0 * SL;
  short* trgxb = bfr + 1 * SL;
  short* x1b   = bfr + 2 * SL;
  short* x2b   = bfr + 3 * SL;
  short* x3b   = bfr + 4 * SL;
  short* qbb   = bfr + 5 * SL;
  short* kbb   = bfr + 6 * SL;
  short* vtb   = bfr + 7 * SL;                          // V^T: E x (B*S)
  short* ctxb  = bfr + 8 * SL;
  short* ffnhb = bfr + 9 * SL;                          // 4*SL (B*S x DF)
  short* watt  = bfr + 13 * SL;                         // 8 x (E*E) attn weights
  short* f1t   = watt + 8 * EEc;                        // DF x E
  short* f2t   = f1t + (long long)E * DF;               // E x DF
  short* fct   = f2t + (long long)DF * E;               // V x E

  dim3 blk(256);

  auto gemm = [&](const short* Ap, long long aO, long long aI, int lda,
                  const short* Bp, long long bO, long long bI, int ldb,
                  float* Cf, short* Cb, int cbT, int ldcb,
                  long long cO, long long cI, int ldc,
                  const float* bias, int M, int N, int K, int inner, int batches,
                  int relu, float alpha) {
    dim3 g((unsigned)((N + BN - 1) / BN), (unsigned)((M + BM - 1) / BM),
           (unsigned)batches);
    gemm_bf16nt_wmma_kernel<<<g, blk, 0, stream>>>(
        Ap, aO, aI, lda, Bp, bO, bI, ldb, Cf, Cb, cbT, ldcb, cO, cI, ldc,
        bias, M, N, K, inner, relu, alpha);
  };
  auto wconv = [&](int wi, short* dst, int Kd, int Nd) {
    wt_transpose_bf16_kernel<<<dim3((unsigned)(Nd / 32), (unsigned)(Kd / 32)),
                               blk, 0, stream>>>(P(wi), dst, Kd, Nd);
  };

  // ---- one-time weight convert + transpose (live decoder layer + fc) ----
  wconv(DB + 0, watt + 0 * EEc, E, E);   // sa.wq
  wconv(DB + 2, watt + 1 * EEc, E, E);   // sa.wk
  wconv(DB + 4, watt + 2 * EEc, E, E);   // sa.wv
  wconv(DB + 6, watt + 3 * EEc, E, E);   // sa.wo
  wconv(DB + 8, watt + 4 * EEc, E, E);   // ca.wq
  wconv(DB + 10, watt + 5 * EEc, E, E);  // ca.wk
  wconv(DB + 12, watt + 6 * EEc, E, E);  // ca.wv
  wconv(DB + 14, watt + 7 * EEc, E, E);  // ca.wo
  wconv(DB + 16, f1t, E, DF);            // f1
  wconv(DB + 18, f2t, DF, E);            // f2
  wconv(89, fct, E, V);                  // fc

  // ---- embeddings + PE (src f32 copy unneeded: encoder stack is dead) ----
  embed_pe_kernel<<<dim3((unsigned)(SL / 256)), blk, 0, stream>>>(
      src, enc_emb, pe, nullptr, srcxb, S, E);
  embed_pe_kernel<<<dim3((unsigned)(SL / 256)), blk, 0, stream>>>(
      trg, dec_emb, pe, trgx, trgxb, S, E);

  // ---- MHA block + residual LN ----
  auto attention = [&](const short* qin, const short* kvin, const short* wt,
                       int pbias, const int* mask_tok, int causal,
                       const float* resid, int lnS, int lnB,
                       float* outF, short* outB) {
    gemm(qin, 0, 0, E, wt + 0 * EEc, 0, 0, E, nullptr, qbb, 0, E,
         0, 0, E, P(pbias + 1), TOKS, E, E, 1, 1, 0, 1.0f);
    gemm(kvin, 0, 0, E, wt + 1 * EEc, 0, 0, E, nullptr, kbb, 0, E,
         0, 0, E, P(pbias + 3), TOKS, E, E, 1, 1, 0, 1.0f);
    gemm(kvin, 0, 0, E, wt + 2 * EEc, 0, 0, E, nullptr, vtb, 1, TOKS,
         0, 0, E, P(pbias + 5), TOKS, E, E, 1, 1, 0, 1.0f);     // V^T store
    // scores[b,h] = Q K^T / 8  (NT, batched over B*H)
    gemm(qbb, SE, DK, E, kbb, SE, DK, E, scores, nullptr, 0, 0,
         (long long)Hh * SS, SS, S, nullptr, S, S, DK, Hh, 4 * Hh, 0, 0.125f);
    softmax_mask_kernel<<<dim3((unsigned)(4 * Hh * S)), blk, 0, stream>>>(
        scores, mask_tok, S, S, Hh, causal);
    // ctx = P @ V  (probs bf16 in-place over scores: pitch 2*S shorts)
    gemm((const short*)scores, (long long)Hh * 2 * SS, 2 * SS, 2 * S,
         vtb, S, (long long)DK * TOKS, TOKS,
         nullptr, ctxb, 0, E, SE, DK, E,
         nullptr, S, DK, S, Hh, 4 * Hh, 0, 1.0f);
    gemm(ctxb, 0, 0, E, wt + 3 * EEc, 0, 0, E, tmp, nullptr, 0, E,
         0, 0, E, P(pbias + 7), TOKS, E, E, 1, 1, 0, 1.0f);
    ln_residual_kernel<<<dim3((unsigned)TOKS), blk, 0, stream>>>(
        tmp, resid, P(lnS), P(lnB), outF, outB);
  };

  // ---- decoder layer (last layer's params; encoder + dec layer 0 are dead
  //      code due to the reference's loop-overwrite bug) ----
  attention(trgxb, trgxb, watt + 0, DB + 0, nullptr, 1,
            trgx, DB + 20, DB + 21, x1, x1b);                   // self (causal)
  attention(x1b, srcxb, watt + 4 * EEc, DB + 8, src, 0,
            x1, DB + 22, DB + 23, x2, x2b);                     // cross (src mask)
  // FFN
  gemm(x2b, 0, 0, E, f1t, 0, 0, E, nullptr, ffnhb, 0, DF,
       0, 0, DF, P(DB + 17), TOKS, DF, E, 1, 1, 1, 1.0f);       // relu
  gemm(ffnhb, 0, 0, DF, f2t, 0, 0, DF, tmp, nullptr, 0, E,
       0, 0, E, P(DB + 19), TOKS, E, DF, 1, 1, 0, 1.0f);
  ln_residual_kernel<<<dim3((unsigned)TOKS), blk, 0, stream>>>(
      tmp, x2, P(DB + 24), P(DB + 25), nullptr, x3b);

  // ---- final FC -> d_out ----
  gemm(x3b, 0, 0, E, fct, 0, 0, E, (float*)d_out, nullptr, 0, V,
       0, 0, V, P(90), TOKS, V, E, 1, 1, 0, 1.0f);
}
